// HeterSumGraph_515396075666
// MI455X (gfx1250) — matrix operand order repaired and checked by reading, not compiled, not run on hardware
//
#include <hip/hip_runtime.h>
#include <math.h>

#define DW    300
#define DSIN  768
#define DH    128
#define NW    50000
#define NS    10000
#define NEDGE 600000
#define NLOOP 10000
#define ETOT  (NEDGE + NLOOP)
#define ENC_NEG_INF 0x007FFFFFu
#define NEG_SLOPE 0.2f

typedef __attribute__((ext_vector_type(16))) __bf16        v16bf;
typedef __attribute__((ext_vector_type(8)))  float         v8f;
typedef __attribute__((ext_vector_type(4)))  unsigned int  u32x4;

union FragBF { u32x4 u[2]; v16bf v; };

__device__ __forceinline__ unsigned short f2bf(float f) {
  unsigned u = __float_as_uint(f);
  unsigned r = u + 0x7fffu + ((u >> 16) & 1u);   // round-to-nearest-even
  return (unsigned short)(r >> 16);
}
__device__ __forceinline__ unsigned pack_bf2(float lo, float hi) {
  return (unsigned)f2bf(lo) | ((unsigned)f2bf(hi) << 16);
}
__device__ __forceinline__ unsigned enc_f(float f) {
  unsigned u = __float_as_uint(f);
  return (u & 0x80000000u) ? ~u : (u | 0x80000000u);
}
__device__ __forceinline__ float dec_f(unsigned u) {
  u = (u & 0x80000000u) ? (u & 0x7fffffffu) : ~u;
  return __uint_as_float(u);
}

// ---------------------------------------------------------------------------
// Out[M,128] = (X [+ pre_add + pre_bias]) @ W[128,K]^T + bias   (bf16 WMMA)
// Block: 256 threads (8 waves) -> 128 rows x 128 cols tile.
// FUSED=true only used with K == DH (final residual GEMMs).
// ---------------------------------------------------------------------------
template<bool FUSED>
__global__ __launch_bounds__(256)
void linear_wmma_kernel(const float* __restrict__ X, const float* __restrict__ W,
                        const float* __restrict__ bias,
                        const float* __restrict__ pre_add,
                        const float* __restrict__ pre_bias,
                        float* __restrict__ Out, int M, int K)
{
  __shared__ __attribute__((aligned(16))) unsigned short lA[128 * 32];
  __shared__ __attribute__((aligned(16))) unsigned short lB[128 * 32];

  const int t        = threadIdx.x;
  const int lane     = t & 31;
  const int wave     = t >> 5;
  const int blockRow = blockIdx.x * 128;

  v8f acc[8] = {};

  const int rowA  = t >> 1;          // 0..127
  const int halfc = (t & 1) * 16;    // 0 or 16
  const int grow  = blockRow + rowA;

  for (int kk = 0; kk < K; kk += 32) {
    const bool fullK = (kk + 32 <= K);
    __syncthreads();

    // ---- stage A tile [128 x 32] ----
    if (fullK && grow < M) {
      const float4* px = (const float4*)(X + (size_t)grow * K + kk + halfc);
      float4 x0 = px[0], x1 = px[1], x2 = px[2], x3 = px[3];
      if (FUSED) {
        const float4* pa = (const float4*)(pre_add + (size_t)grow * K + kk + halfc);
        const float4* pb = (const float4*)(pre_bias + kk + halfc);
        float4 a0 = pa[0], a1 = pa[1], a2 = pa[2], a3 = pa[3];
        float4 b0 = pb[0], b1 = pb[1], b2 = pb[2], b3 = pb[3];
        x0.x += a0.x + b0.x; x0.y += a0.y + b0.y; x0.z += a0.z + b0.z; x0.w += a0.w + b0.w;
        x1.x += a1.x + b1.x; x1.y += a1.y + b1.y; x1.z += a1.z + b1.z; x1.w += a1.w + b1.w;
        x2.x += a2.x + b2.x; x2.y += a2.y + b2.y; x2.z += a2.z + b2.z; x2.w += a2.w + b2.w;
        x3.x += a3.x + b3.x; x3.y += a3.y + b3.y; x3.z += a3.z + b3.z; x3.w += a3.w + b3.w;
      }
      u32x4* dst16 = (u32x4*)&lA[rowA * 32 + halfc];
      u32x4 p0 = { pack_bf2(x0.x, x0.y), pack_bf2(x0.z, x0.w),
                   pack_bf2(x1.x, x1.y), pack_bf2(x1.z, x1.w) };
      u32x4 p1 = { pack_bf2(x2.x, x2.y), pack_bf2(x2.z, x2.w),
                   pack_bf2(x3.x, x3.y), pack_bf2(x3.z, x3.w) };
      dst16[0] = p0; dst16[1] = p1;
    } else {
      #pragma unroll
      for (int j = 0; j < 16; ++j) {
        int c = halfc + j;
        float v = 0.f;
        if (grow < M && (kk + c) < K) {
          v = X[(size_t)grow * K + kk + c];
          if (FUSED) v += pre_add[(size_t)grow * K + kk + c] + pre_bias[kk + c];
        }
        lA[rowA * 32 + c] = f2bf(v);
      }
    }

    // ---- stage B tile: B[k][n] = W[n][k]; stored [n][32] ----
    if (fullK) {
      const float4* pw = (const float4*)(W + (size_t)rowA * K + kk + halfc);
      float4 w0 = pw[0], w1 = pw[1], w2 = pw[2], w3 = pw[3];
      u32x4* dst16 = (u32x4*)&lB[rowA * 32 + halfc];
      u32x4 p0 = { pack_bf2(w0.x, w0.y), pack_bf2(w0.z, w0.w),
                   pack_bf2(w1.x, w1.y), pack_bf2(w1.z, w1.w) };
      u32x4 p1 = { pack_bf2(w2.x, w2.y), pack_bf2(w2.z, w2.w),
                   pack_bf2(w3.x, w3.y), pack_bf2(w3.z, w3.w) };
      dst16[0] = p0; dst16[1] = p1;
    } else {
      #pragma unroll
      for (int j = 0; j < 16; ++j) {
        int c = halfc + j;
        float v = ((kk + c) < K) ? W[(size_t)rowA * K + kk + c] : 0.f;
        lB[rowA * 32 + c] = f2bf(v);
      }
    }
    __syncthreads();

    // A fragment: lane<16 -> M=lane, halves K 0..7 | 16..23 ; lane>=16 -> K 8..15 | 24..31
    FragBF fa;
    {
      const unsigned short* pr = &lA[(wave * 16 + (lane & 15)) * 32];
      const int k0 = (lane & 16) ? 8 : 0;
      fa.u[0] = *(const u32x4*)(pr + k0);
      fa.u[1] = *(const u32x4*)(pr + k0 + 16);
    }
    // Preload all 8 B fragments, then issue WMMAs back-to-back.
    FragBF fb[8];
    {
      const unsigned short* pc0 = &lB[(lane & 15) * 32 + ((lane & 16) ? 8 : 0)];
      #pragma unroll
      for (int nt = 0; nt < 8; ++nt) {
        const unsigned short* pc = pc0 + nt * (16 * 32);
        fb[nt].u[0] = *(const u32x4*)(pc);
        fb[nt].u[1] = *(const u32x4*)(pc + 16);
      }
    }
    #pragma unroll
    for (int nt = 0; nt < 8; ++nt) {
      acc[nt] = __builtin_amdgcn_wmma_f32_16x16x32_bf16(
          false, fa.v, false, fb[nt].v, (short)0, acc[nt], false, false);
    }
  }

  // Epilogue: C layout -> VGPR j: lanes 0-15 row=j, lanes 16-31 row=j+8; col = lane&15
  const int colLo  = lane & 15;
  const int rHiOff = (lane & 16) ? 8 : 0;
  const int rowBase = blockRow + wave * 16 + rHiOff;
  if (blockRow + 128 <= M) {
    #pragma unroll
    for (int nt = 0; nt < 8; ++nt) {
      int col = nt * 16 + colLo;
      float b = bias ? bias[col] : 0.f;
      #pragma unroll
      for (int j = 0; j < 8; ++j)
        Out[(size_t)(rowBase + j) * DH + col] = acc[nt][j] + b;
    }
  } else {
    #pragma unroll
    for (int nt = 0; nt < 8; ++nt) {
      int col = nt * 16 + colLo;
      float b = bias ? bias[col] : 0.f;
      #pragma unroll
      for (int j = 0; j < 8; ++j) {
        int row = rowBase + j;
        if (row < M) Out[(size_t)row * DH + col] = acc[nt][j] + b;
      }
    }
  }
}

// out[row] = dot(H[row,:128], avec)  -- one wave per row
__global__ __launch_bounds__(256)
void rowdot_kernel(const float* __restrict__ H, const float* __restrict__ avec,
                   float* __restrict__ out, int M)
{
  int row  = blockIdx.x * 8 + (threadIdx.x >> 5);
  int lane = threadIdx.x & 31;
  if (row >= M) return;
  const float4 h = ((const float4*)(H + (size_t)row * DH))[lane];
  const float4 a = ((const float4*)avec)[lane];
  float s = h.x * a.x + h.y * a.y + h.z * a.z + h.w * a.w;
  #pragma unroll
  for (int m = 16; m >= 1; m >>= 1) s += __shfl_xor(s, m, 32);
  if (lane == 0) out[row] = s;
}

__global__ __launch_bounds__(256)
void fill_u32_kernel(unsigned* __restrict__ p, unsigned v, long long n)
{
  long long i = (long long)blockIdx.x * blockDim.x + threadIdx.x;
  if (i < n) p[i] = v;
}

// pass 1: alpha = leaky_relu(as[s] + ad[d]); mask numeric self loops; atomic max per dst
__global__ __launch_bounds__(256)
void edge_alpha_max_kernel(const int* __restrict__ src, const int* __restrict__ dst,
                           const float* __restrict__ as_, const float* __restrict__ ad_,
                           float* __restrict__ alpha, unsigned* __restrict__ mmax)
{
  int i = blockIdx.x * blockDim.x + threadIdx.x;
  if (i >= ETOT) return;
  int s, d; bool valid;
  if (i < NEDGE) { s = src[i]; d = dst[i]; valid = (s != d); }
  else           { s = d = i - NEDGE;      valid = true; }
  if (valid) {
    float a = as_[s] + ad_[d];
    a = (a > 0.f) ? a : a * NEG_SLOPE;
    alpha[i] = a;
    atomicMax(&mmax[d], enc_f(a));
  } else {
    alpha[i] = -INFINITY;
  }
}

// pass 2: e = exp(alpha - m[d]); atomic sum per dst
__global__ __launch_bounds__(256)
void edge_exp_sum_kernel(const int* __restrict__ dst,
                         const float* __restrict__ alpha, const unsigned* __restrict__ mmax,
                         float* __restrict__ ebuf, float* __restrict__ denom)
{
  int i = blockIdx.x * blockDim.x + threadIdx.x;
  if (i >= ETOT) return;
  int d = (i < NEDGE) ? dst[i] : (i - NEDGE);
  float a = alpha[i];
  float e = 0.f;
  if (a != -INFINITY) {
    float m = dec_f(mmax[d]);
    if (!(m > -INFINITY)) m = 0.f;
    e = __expf(a - m);
    atomicAdd(&denom[d], e);
  }
  ebuf[i] = e;
}

// pass 3: agg[d] += (e/denom[d]) * hs[s]   (one wave per edge, 4 floats/lane)
__global__ __launch_bounds__(256)
void edge_aggregate_kernel(const int* __restrict__ src, const int* __restrict__ dst,
                           const float* __restrict__ hs, const float* __restrict__ ebuf,
                           const float* __restrict__ denom, float* __restrict__ agg)
{
  long long gid = (long long)blockIdx.x * blockDim.x + threadIdx.x;
  int edge = (int)(gid >> 5);
  int lane = (int)(gid & 31);
  if (edge >= ETOT) return;
  int s, d;
  if (edge < NEDGE) { s = src[edge]; d = dst[edge]; }
  else              { s = d = edge - NEDGE; }
  float e = ebuf[edge];
  if (e == 0.f) return;
  float w = e / fmaxf(denom[d], 1e-16f);
  const float4 h = ((const float4*)(hs + (size_t)s * DH))[lane];
  float* o = agg + (size_t)d * DH + lane * 4;
  atomicAdd(o + 0, w * h.x);
  atomicAdd(o + 1, w * h.y);
  atomicAdd(o + 2, w * h.z);
  atomicAdd(o + 3, w * h.w);
}

extern "C" void kernel_launch(void* const* d_in, const int* in_sizes, int n_in,
                              void* d_out, int out_size, void* d_ws, size_t ws_size,
                              hipStream_t stream)
{
  const float* Xw    = (const float*)d_in[0];
  const float* Xs    = (const float*)d_in[1];
  const int*   E     = (const int*)  d_in[2];
  const float* W1    = (const float*)d_in[3];
  const float* b1    = (const float*)d_in[4];
  const float* W2    = (const float*)d_in[5];
  const float* b2    = (const float*)d_in[6];
  const float* W3    = (const float*)d_in[7];
  const float* b3    = (const float*)d_in[8];
  const float* W4    = (const float*)d_in[9];
  const float* b4    = (const float*)d_in[10];
  const float* g1_Ws = (const float*)d_in[11];
  const float* g1_Wd = (const float*)d_in[12];
  const float* g1_as = (const float*)d_in[13];
  const float* g1_ad = (const float*)d_in[14];
  const float* g1_b  = (const float*)d_in[15];
  const float* g2_Ws = (const float*)d_in[16];
  const float* g2_Wd = (const float*)d_in[17];
  const float* g2_as = (const float*)d_in[18];
  const float* g2_ad = (const float*)d_in[19];
  const float* g2_b  = (const float*)d_in[20];

  const int* src = E;          // E[0]
  const int* dst = E + NEDGE;  // E[1]

  float* p = (float*)d_ws;
  float* Hw  = p; p += (size_t)NW * DH;
  float* Hs  = p; p += (size_t)NS * DH;
  float* T1  = p; p += (size_t)NW * DH;   // hd2 -> aggW
  float* T2  = p; p += (size_t)NS * DH;   // hs2 -> aggS
  float* T3  = p; p += (size_t)NS * DH;   // hd1
  float* T4  = p; p += (size_t)NW * DH;   // hs1
  float* as2 = p; p += NS;
  float* ad2 = p; p += NW;
  float* as1 = p; p += NW;
  float* ad1 = p; p += NS;
  unsigned* mmaxW = (unsigned*)p; p += NW;
  float*    denomW = p;          p += NW;
  unsigned* mmaxS = (unsigned*)p; p += NS;
  float*    denomS = p;          p += NS;
  float* alpha = p; p += ETOT;
  float* ebuf  = p; p += ETOT;

  auto gblk = [](int M) { return dim3((unsigned)((M + 127) / 128)); };
  auto gfil = [](long long n) { return dim3((unsigned)((n + 255) / 256)); };
  const dim3 B256(256);
  const dim3 gE((ETOT + 255) / 256);
  const dim3 gAgg((unsigned)(((long long)ETOT * 32 + 255) / 256));

  // Input projections
  linear_wmma_kernel<false><<<gblk(NW), B256, 0, stream>>>(Xw, W1, b1, nullptr, nullptr, Hw, NW, DW);
  linear_wmma_kernel<false><<<gblk(NS), B256, 0, stream>>>(Xs, W2, b2, nullptr, nullptr, Hs, NS, DSIN);

  // ---- GAT2: src feats = Hs, dst feats = Hw -> aggW ----
  linear_wmma_kernel<false><<<gblk(NW), B256, 0, stream>>>(Hw, g2_Wd, nullptr, nullptr, nullptr, T1, NW, DH); // hd2
  rowdot_kernel<<<dim3((NW + 7) / 8), B256, 0, stream>>>(T1, g2_ad, ad2, NW);
  linear_wmma_kernel<false><<<gblk(NS), B256, 0, stream>>>(Hs, g2_Ws, nullptr, nullptr, nullptr, T2, NS, DH); // hs2
  rowdot_kernel<<<dim3((NS + 7) / 8), B256, 0, stream>>>(T2, g2_as, as2, NS);
  fill_u32_kernel<<<gfil(NW), B256, 0, stream>>>(mmaxW, ENC_NEG_INF, NW);
  fill_u32_kernel<<<gfil(NW), B256, 0, stream>>>((unsigned*)denomW, 0u, NW);
  edge_alpha_max_kernel<<<gE, B256, 0, stream>>>(src, dst, as2, ad2, alpha, mmaxW);
  edge_exp_sum_kernel<<<gE, B256, 0, stream>>>(dst, alpha, mmaxW, ebuf, denomW);
  fill_u32_kernel<<<gfil((long long)NW * DH), B256, 0, stream>>>((unsigned*)T1, 0u, (long long)NW * DH); // aggW
  edge_aggregate_kernel<<<gAgg, B256, 0, stream>>>(src, dst, T2, ebuf, denomW, T1);

  // ---- GAT1: src feats = Hw, dst feats = Hs -> aggS ----
  linear_wmma_kernel<false><<<gblk(NS), B256, 0, stream>>>(Hs, g1_Wd, nullptr, nullptr, nullptr, T3, NS, DH); // hd1
  rowdot_kernel<<<dim3((NS + 7) / 8), B256, 0, stream>>>(T3, g1_ad, ad1, NS);
  linear_wmma_kernel<false><<<gblk(NW), B256, 0, stream>>>(Hw, g1_Ws, nullptr, nullptr, nullptr, T4, NW, DH); // hs1
  rowdot_kernel<<<dim3((NW + 7) / 8), B256, 0, stream>>>(T4, g1_as, as1, NW);
  fill_u32_kernel<<<gfil(NS), B256, 0, stream>>>(mmaxS, ENC_NEG_INF, NS);
  fill_u32_kernel<<<gfil(NS), B256, 0, stream>>>((unsigned*)denomS, 0u, NS);
  edge_alpha_max_kernel<<<gE, B256, 0, stream>>>(src, dst, as1, ad1, alpha, mmaxS);
  edge_exp_sum_kernel<<<gE, B256, 0, stream>>>(dst, alpha, mmaxS, ebuf, denomS);
  fill_u32_kernel<<<gfil((long long)NS * DH), B256, 0, stream>>>((unsigned*)T2, 0u, (long long)NS * DH); // aggS
  edge_aggregate_kernel<<<gAgg, B256, 0, stream>>>(src, dst, T4, ebuf, denomS, T2);

  // Final linears: Out = (agg + H + g_b) @ W^T + b   (residual fused into A-staging)
  float* OutW = (float*)d_out;
  float* OutS = OutW + (size_t)NW * DH;
  linear_wmma_kernel<true><<<gblk(NW), B256, 0, stream>>>(T1, W4, b4, Hw, g2_b, OutW, NW, DH);
  linear_wmma_kernel<true><<<gblk(NS), B256, 0, stream>>>(T2, W3, b3, Hs, g1_b, OutS, NS, DH);
}
